// fbMnist_14989435863734
// MI455X (gfx1250) — compile-verified, matrix operand order checked by reading, not run on hardware
//
#include <hip/hip_runtime.h>

// ---------------------------------------------------------------------------
// Recurrent spiking MLP (LIF) on gfx1250 — round 3.
//   grid = B/16 workgroups (batch tiles), 128 threads (4 waves) each.
//   Weight WMMA B-fragments hoisted to VGPRs (>256 VGPRs via s_set_vgpr_msb).
//   All WMMA chains start from inline-0 C; biases folded into the LIF update
//   (v' = fma(v,0.5,0.5*bias) + 0.5*gemm), b3 added once at the end (784*b3).
//   ONE barrier per timestep:
//     phase1: stage x(t+1), L1 GEMM (K=96 fused [x|s1rec]), LIF1, spike writes
//     barrier
//     phase2: L2 GEMM (K=320 fused [s1|s2rec]), LIF2, spike writes,
//             L3 partial GEMM (per-wave K-chunk of 64, same-wave LDS reuse)
//   Membrane potentials + output accumulator live in VGPRs for all 784 steps.
// ---------------------------------------------------------------------------

typedef __attribute__((ext_vector_type(16))) _Float16 v16h;
typedef __attribute__((ext_vector_type(2)))  _Float16 h2;
typedef __attribute__((ext_vector_type(8)))  float    v8f;

#define LDK1 96    // K layer1:  8 (x) + 64 (s1 rec) + 24 pad
#define LDK2 320   // K layer2: 64 (s1) + 256 (s2 rec)
#define LDK3 256   // K layer3: 256 (s2)

// shared-memory layout, offsets in _Float16 units
#define OFF_WC1  0                               // [64][96]
#define OFF_WC2  (OFF_WC1 + 64 * LDK1)           // [256][320]
#define OFF_W3P  (OFF_WC2 + 256 * LDK2)          // [16][256]
#define OFF_B1   (OFF_W3P + 16 * LDK3)           // 2 x [16][96]
#define OFF_B2   (OFF_B1 + 2 * 16 * LDK1)        // 2 x [16][320]
#define OFF_XT   (OFF_B2 + 2 * 16 * LDK2)        // [16][784] staged input (f16)
#define OFF_BIAS (OFF_XT + 16 * 784)             // floats: 64 + 256 + 16
#define SMEM_HALVES (OFF_BIAS + 2 * (64 + 256 + 16))
#define SMEM_BYTES  (SMEM_HALVES * 2)            // 237376 B < 320 KB

// Load one 16x32 f16 WMMA operand fragment from LDS (A: row=M=lane&15;
// B: row = weight row = output col = lane&15, since B[k][n] == W[n][k]).
// 16-bit layout: lane<16 holds K={0..7,16..23}, lane>=16 holds K={8..15,24..31}.
__device__ __forceinline__ v16h load_frag(const _Float16* p, int row, int ldk,
                                          int k0, int hi) {
  union { v16h v; h2 p2[8]; } f;
  const _Float16* base = p + row * ldk + k0 + hi * 8;
#pragma unroll
  for (int t = 0; t < 8; ++t) {
    const int kp = 2 * t + (t >= 4 ? 8 : 0);
    f.p2[t] = *(const h2*)(base + kp);
  }
  return f.v;
}

__device__ __forceinline__ v8f wmma_f16(v16h a, v16h b, v8f c) {
  return __builtin_amdgcn_wmma_f32_16x16x32_f16(false, a, false, b,
                                                (short)0, c, false, false);
}

__global__ void __launch_bounds__(128, 1)
snn_lif_kernel(const float* __restrict__ x,
               const float* __restrict__ W1,  const float* __restrict__ b1,
               const float* __restrict__ Wr1, const float* __restrict__ br1,
               const float* __restrict__ W2,  const float* __restrict__ b2,
               const float* __restrict__ Wr2, const float* __restrict__ br2,
               const float* __restrict__ W3,  const float* __restrict__ b3,
               float* __restrict__ out) {
  extern __shared__ _Float16 smem[];
  _Float16* WC1 = smem + OFF_WC1;   // [W1 | Wr1 | 0] as [64][96]
  _Float16* WC2 = smem + OFF_WC2;   // [W2 | Wr2]     as [256][320]
  _Float16* W3P = smem + OFF_W3P;   // W3 padded to   [16][256]
  _Float16* B1  = smem + OFF_B1;    // A-panel L1 (x | s1), double buffered
  _Float16* B2  = smem + OFF_B2;    // A-panel L2 (s1 | s2), double buffered
  _Float16* XT  = smem + OFF_XT;    // full input tile [16][784]
  float* biasA  = (float*)(smem + OFF_BIAS);
  float* biasB  = biasA + 64;
  float* b3f    = biasB + 256;

  const int tid  = threadIdx.x;
  const int wg   = blockIdx.x;
  const int nthr = blockDim.x;       // 128

  // ---- one-time setup: f32->f16 weight concat into LDS ----
  for (int i = tid; i < 64 * LDK1; i += nthr) {
    const int n = i / LDK1, k = i % LDK1;
    const float w = (k < 8) ? W1[n * 8 + k]
                  : (k < 72) ? Wr1[n * 64 + (k - 8)] : 0.0f;
    WC1[i] = (_Float16)w;
  }
  for (int i = tid; i < 256 * LDK2; i += nthr) {
    const int n = i / LDK2, k = i % LDK2;
    const float w = (k < 64) ? W2[n * 64 + k] : Wr2[n * 256 + (k - 64)];
    WC2[i] = (_Float16)w;
  }
  for (int i = tid; i < 16 * LDK3; i += nthr) {
    const int n = i / LDK3, k = i % LDK3;
    W3P[i] = (_Float16)((n < 10) ? W3[n * 256 + k] : 0.0f);
  }
  for (int i = tid; i < 2 * 16 * LDK1; i += nthr) B1[i] = (_Float16)0.0f;
  for (int i = tid; i < 2 * 16 * LDK2; i += nthr) B2[i] = (_Float16)0.0f;
  for (int i = tid; i < 16 * 784; i += nthr) {
    const int m = i / 784, t = i % 784;
    XT[i] = (_Float16)x[(wg * 16 + m) * 784 + t];
  }
  for (int i = tid; i < 64;  i += nthr) biasA[i] = b1[i] + br1[i];
  for (int i = tid; i < 256; i += nthr) biasB[i] = b2[i] + br2[i];
  for (int i = tid; i < 16;  i += nthr) b3f[i] = (i < 10) ? b3[i] : 0.0f;
  // stage x(0) into B1[0] cols 0..7 directly from global
  {
    const int mm = tid >> 3, kk = tid & 7;   // 128 threads = 16x8 exactly
    smem[OFF_B1 + mm * LDK1 + kk] = (_Float16)x[(wg * 16 + mm) * 784 + kk];
  }
  __syncthreads();

  const int lane = tid & 31;
  const int wave = tid >> 5;
  const int col  = lane & 15;   // N for B/C/D, M for A
  const int hi   = lane >> 4;
  const int m0   = hi * 8;      // C/D row base for this lane

  // ---- hoist all loop-invariant weight B-fragments into VGPRs ----
  const int n0 = wave * 16;                 // this wave's L1 tile
  v16h bf1[3];
#pragma unroll
  for (int kc = 0; kc < 3; ++kc)
    bf1[kc] = load_frag(WC1, n0 + col, LDK1, kc * 32, hi);
  v16h bf2[10][4];
#pragma unroll
  for (int kc = 0; kc < 10; ++kc)
#pragma unroll
    for (int j = 0; j < 4; ++j)
      bf2[kc][j] = load_frag(WC2, wave * 64 + j * 16 + col, LDK2, kc * 32, hi);
  v16h bf3[2];                              // L3 K-chunk owned by this wave
#pragma unroll
  for (int kc = 0; kc < 2; ++kc)
    bf3[kc] = load_frag(W3P, col, LDK3, wave * 64 + kc * 32, hi);

  // register-resident LIF state + per-wave partial output accumulator
  float v1s[8], v2s[4][8], accv[8];
#pragma unroll
  for (int r = 0; r < 8; ++r) { v1s[r] = 0.0f; accv[r] = 0.0f; }
#pragma unroll
  for (int j = 0; j < 4; ++j)
#pragma unroll
    for (int r = 0; r < 8; ++r) v2s[j][r] = 0.0f;

  // half-biases folded into the LIF update: v' = fma(v,0.5,hb) + 0.5*gemm
  const float hb1 = 0.5f * biasA[n0 + col];
  float hb2[4];
#pragma unroll
  for (int j = 0; j < 4; ++j) hb2[j] = 0.5f * biasB[wave * 64 + j * 16 + col];

  // ---- sequential time loop: ONE barrier per step ----
  for (int t = 0; t < 784; ++t) {
    const int cur = t & 1, nxt = cur ^ 1;
    _Float16* B1c = B1 + cur * (16 * LDK1);
    _Float16* B1n = B1 + nxt * (16 * LDK1);
    _Float16* B2c = B2 + cur * (16 * LDK2);
    _Float16* B2n = B2 + nxt * (16 * LDK2);

    // phase 1a: stage x(t+1) for next step (B1n cols 0..7; disjoint from s1)
    {
      const int mm = tid >> 3, kk = tid & 7;
      const int tn = t + 1;
      const int start = (tn < 776) ? tn : 776;
      B1n[mm * LDK1 + kk] = XT[mm * 784 + start + kk];
    }

    // phase 1b: layer 1  [x_t | s1_prev] @ [W1|Wr1]^T, K=96, 1 tile/wave
    v8f c1 = {};                              // inline-0 C operand
#pragma unroll
    for (int kc = 0; kc < 3; ++kc) {
      const v16h a = load_frag(B1c, col, LDK1, kc * 32, hi);
      c1 = wmma_f16(a, bf1[kc], c1);
    }
#pragma unroll
    for (int r = 0; r < 8; ++r) {             // LIF layer 1 (bias folded)
      const int mm = r + m0;
      float vv = __builtin_fmaf(v1s[r], 0.5f, hb1);
      vv = __builtin_fmaf(c1[r], 0.5f, vv);   // v' = 0.5 v + 0.5(gemm+bias)
      const float s = (vv >= 1.0f) ? 1.0f : 0.0f;
      vv *= (1.0f - s);                       // hard reset
      v1s[r] = vv;
      const _Float16 sh = (_Float16)s;
      B1n[mm * LDK1 + 8 + n0 + col] = sh;     // next step's L1 recurrence
      B2c[mm * LDK2 + n0 + col]     = sh;     // this step's L2 input
    }
    __syncthreads();                          // the only barrier per step

    // phase 2a: layer 2  [s1 | s2_prev] @ [W2|Wr2]^T, K=320, 4 tiles/wave
    v8f c2[4] = {{}, {}, {}, {}};
#pragma unroll
    for (int kc = 0; kc < 10; ++kc) {
      const v16h a = load_frag(B2c, col, LDK2, kc * 32, hi);
#pragma unroll
      for (int j = 0; j < 4; ++j)
        c2[j] = wmma_f16(a, bf2[kc][j], c2[j]);
    }
#pragma unroll
    for (int j = 0; j < 4; ++j)
#pragma unroll
      for (int r = 0; r < 8; ++r) {           // LIF layer 2 (bias folded)
        const int mm = r + m0;
        float vv = __builtin_fmaf(v2s[j][r], 0.5f, hb2[j]);
        vv = __builtin_fmaf(c2[j][r], 0.5f, vv);
        const float s = (vv >= 1.0f) ? 1.0f : 0.0f;
        vv *= (1.0f - s);
        v2s[j][r] = vv;
        B2n[mm * LDK2 + 64 + wave * 64 + j * 16 + col] = (_Float16)s;
      }

    // phase 2b: layer 3 partial — this wave's K-chunk of s2 @ W3^T.
    // Reads only this wave's own freshly written s2 columns: same-wave LDS
    // ops are in-order, so no barrier is needed. b3 is added once at the end.
    {
      v8f c3 = {};
#pragma unroll
      for (int kc = 0; kc < 2; ++kc) {
        const v16h a = load_frag(B2n + 64 + wave * 64, col, LDK2, kc * 32, hi);
        c3 = wmma_f16(a, bf3[kc], c3);
      }
#pragma unroll
      for (int r = 0; r < 8; ++r) accv[r] += c3[r];
    }
  }

  // ---- reduce 4 per-wave partials + T*b3, write acc [16][10] ----
  float* accbuf = (float*)(smem + OFF_B2);   // reuse dead LDS: 4*16*16 floats
#pragma unroll
  for (int r = 0; r < 8; ++r)
    accbuf[wave * 256 + (r + m0) * 16 + col] = accv[r];
  __syncthreads();
  if (wave == 0 && col < 10) {
    const float tb3 = 784.0f * b3f[col];     // per-step b3, summed over T
#pragma unroll
    for (int r = 0; r < 8; ++r) {
      const int mm = r + m0;
      const float s = accbuf[mm * 16 + col] + accbuf[256 + mm * 16 + col] +
                      accbuf[512 + mm * 16 + col] + accbuf[768 + mm * 16 + col] +
                      tb3;
      out[(wg * 16 + mm) * 10 + col] = s;
    }
  }
}

extern "C" void kernel_launch(void* const* d_in, const int* in_sizes, int n_in,
                              void* d_out, int out_size, void* d_ws, size_t ws_size,
                              hipStream_t stream) {
  (void)n_in; (void)d_ws; (void)ws_size; (void)out_size;
  const float* x   = (const float*)d_in[0];
  const float* W1  = (const float*)d_in[1];
  const float* b1  = (const float*)d_in[2];
  const float* Wr1 = (const float*)d_in[3];
  const float* br1 = (const float*)d_in[4];
  const float* W2  = (const float*)d_in[5];
  const float* b2  = (const float*)d_in[6];
  const float* Wr2 = (const float*)d_in[7];
  const float* br2 = (const float*)d_in[8];
  const float* W3  = (const float*)d_in[9];
  const float* b3  = (const float*)d_in[10];
  float* out = (float*)d_out;

  const int B = in_sizes[0] / 784;          // 256
  const int nblocks = B / 16;               // 16 workgroups, one per batch tile

  (void)hipFuncSetAttribute((const void*)snn_lif_kernel,
                            hipFuncAttributeMaxDynamicSharedMemorySize,
                            SMEM_BYTES);

  snn_lif_kernel<<<dim3(nblocks), dim3(128), SMEM_BYTES, stream>>>(
      x, W1, b1, Wr1, br1, W2, b2, Wr2, br2, W3, b3, out);
}
